// EGNN_2688649527658
// MI455X (gfx1250) — matrix operand order repaired and checked by reading, not compiled, not run on hardware
//
#include <hip/hip_runtime.h>
#include <hip/hip_bf16.h>

typedef __attribute__((ext_vector_type(16))) _Float16 v16h;
typedef __attribute__((ext_vector_type(8)))  _Float16 v8h;
typedef __attribute__((ext_vector_type(8)))  float    v8f;

#define DF 128   // feature dim
#define WPB 4    // waves per block for the WMMA kernels

// ---------- helpers ----------
__device__ __forceinline__ v16h load_frag16(const _Float16* p) {
    v8h a = *(const v8h*)p;
    v8h b = *(const v8h*)(p + 8);
    v16h r;
#pragma unroll
    for (int i = 0; i < 8; ++i) { r[i] = a[i]; r[i + 8] = b[i]; }
    return r;
}

// ---------- small utility kernels ----------
__global__ void zero_f32(float* p, int n) {
    int i = blockIdx.x * blockDim.x + threadIdx.x;
    if (i < n) p[i] = 0.f;
}

__global__ void init_x(const float* __restrict__ x, _Float16* __restrict__ xh,
                       float* __restrict__ xb, int n) {
    int i = blockIdx.x * blockDim.x + threadIdx.x;
    if (i < n) { float v = x[i]; xh[i] = (_Float16)v; xb[i] = v; }
}

__global__ void f32_to_f16(const float* __restrict__ a, _Float16* __restrict__ b, int n) {
    int i = blockIdx.x * blockDim.x + threadIdx.x;
    if (i < n) b[i] = (_Float16)a[i];
}

__global__ void count_kernel(const long long* __restrict__ dst, float* __restrict__ cnt, int E) {
    int i = blockIdx.x * blockDim.x + threadIdx.x;
    if (i < E) atomicAdd(&cnt[(int)dst[i]], 1.0f);
}

__global__ void invden_kernel(const float* __restrict__ cnt, float* __restrict__ inv, int N) {
    int i = blockIdx.x * blockDim.x + threadIdx.x;
    if (i < N) inv[i] = 1.0f / fmaxf(cnt[i], 1.0f);
}

// Repack f32 row-major weight W[K][128] into WMMA B-fragment order (f16).
// Fragment (nt,kt): element(lane,e) = W[kt*32 + (e>=8?16:0) + (lane>=16?8:0) + (e&7)][nt*16 + (lane&15)]
// stored at out[(((nt*NK)+kt)*32 + lane)*16 + e]; 16 contiguous halves per lane.
__global__ void repack_b(const float* __restrict__ W, _Float16* __restrict__ out, int NK) {
    int idx = blockIdx.x * blockDim.x + threadIdx.x;
    int total = NK * 8 * 32 * 16;
    if (idx >= total) return;
    int e    = idx & 15;
    int lane = (idx >> 4) & 31;
    int f    = idx >> 9;
    int kt   = f % NK;
    int nt   = f / NK;
    int k = kt * 32 + ((e >> 3) << 4) + ((lane >> 4) << 3) + (e & 7);
    int n = nt * 16 + (lane & 15);
    out[idx] = (_Float16)W[k * DF + n];
}

// ---------- edge kernel: one wave32 per 32-edge tile (two 16-row sub-tiles) ----------
__global__ __launch_bounds__(32 * WPB, 1)
void edge_kernel(const _Float16* __restrict__ xh,
                 const long long* __restrict__ srcI,
                 const long long* __restrict__ dstI,
                 const float* __restrict__ ewgt,
                 const _Float16* __restrict__ w1p,    // packed ew1[0:256], NK=8
                 const float* __restrict__ w1last,    // ew1 row 256 (f32)
                 const float* __restrict__ b1,
                 const _Float16* __restrict__ w2p,    // packed ew2, NK=4
                 const float* __restrict__ b2,
                 const float* __restrict__ invden,
                 float* __restrict__ aggr,
                 int nE)
{
    __shared__ __align__(16) _Float16 mt[WPB][32][DF];
    const int w    = threadIdx.x >> 5;
    const int lane = threadIdx.x & 31;
    const int tile = blockIdx.x * WPB + w;
    const int e0   = tile * 32;
    if (e0 >= nE) return;
    const int hi = lane >> 4;
    const int lr = lane & 15;

    // ---- A fragments for two sub-tiles: gather [x_dst | x_src] in WMMA A layout
    v16h A[2][8];
#pragma unroll
    for (int s = 0; s < 2; ++s) {
        const int eA = min(e0 + s * 16 + lr, nE - 1);
        const _Float16* pi = xh + (size_t)dstI[eA] * DF;
        const _Float16* pj = xh + (size_t)srcI[eA] * DF;
#pragma unroll
        for (int kt = 0; kt < 8; ++kt) {
            const int k0 = kt * 32;
            const _Float16* base = (k0 < DF) ? (pi + k0) : (pj + (k0 - DF));
            v8h lo = *(const v8h*)(base + hi * 8);
            v8h hh = *(const v8h*)(base + 16 + hi * 8);
#pragma unroll
            for (int i = 0; i < 8; ++i) { A[s][kt][i] = lo[i]; A[s][kt][8 + i] = hh[i]; }
        }
    }

    // per-row metadata; D-frag VGPR r <-> row (r + 8*hi) of sub-tile s
    float ewr[2][8]; int drow[2][8]; float dinv[2][8];
#pragma unroll
    for (int s = 0; s < 2; ++s)
#pragma unroll
        for (int r = 0; r < 8; ++r) {
            int ee = min(e0 + s * 16 + r + 8 * hi, nE - 1);
            ewr[s][r] = ewgt[ee];
            int dd = (int)dstI[ee];
            drow[s][r] = dd;
            dinv[s][r] = invden[dd];
        }

    // ---- GEMM1: m = relu([x_i|x_j] @ W1 + ea*w1last + b1)
    // B-fragment stream is software-pipelined: fragment f+1 loads while f computes.
    {
        const _Float16* wbase = w1p + (size_t)lane * 16;
        v16h b = load_frag16(wbase);            // fragment 0
#pragma unroll
        for (int nt = 0; nt < 8; ++nt) {
            v8f c0 = {}, c1 = {};
#pragma unroll
            for (int kt = 0; kt < 8; ++kt) {
                v16h cur = b;
                const int nf = nt * 8 + kt + 1;
                if (nf < 64) b = load_frag16(wbase + (size_t)nf * 32 * 16);
                c0 = __builtin_amdgcn_wmma_f32_16x16x32_f16(false, A[0][kt], false, cur,
                                                            (short)0, c0, false, false);
                c1 = __builtin_amdgcn_wmma_f32_16x16x32_f16(false, A[1][kt], false, cur,
                                                            (short)0, c1, false, false);
            }
            const int n = nt * 16 + lr;
            const float b1n = b1[n];
            const float wln = w1last[n];
#pragma unroll
            for (int r = 0; r < 8; ++r) {
                mt[w][r + 8 * hi][n]      = (_Float16)fmaxf(c0[r] + b1n + ewr[0][r] * wln, 0.f);
                mt[w][16 + r + 8 * hi][n] = (_Float16)fmaxf(c1[r] + b1n + ewr[1][r] * wln, 0.f);
            }
        }
    }
    asm volatile("" ::: "memory");   // order LDS stores before LDS loads (same wave)

    // ---- reload m as A fragments from LDS
    v16h A2[2][4];
#pragma unroll
    for (int s = 0; s < 2; ++s)
#pragma unroll
        for (int kt = 0; kt < 4; ++kt) {
            const _Float16* mrow = &mt[w][s * 16 + lr][kt * 32 + hi * 8];
            v8h lo = *(const v8h*)(mrow);
            v8h hh = *(const v8h*)(mrow + 16);
#pragma unroll
            for (int i = 0; i < 8; ++i) { A2[s][kt][i] = lo[i]; A2[s][kt][8 + i] = hh[i]; }
        }

    // ---- GEMM2: relu(m @ W2 + b2), scaled mean-aggregate via atomics
    {
        const _Float16* wbase = w2p + (size_t)lane * 16;
        v16h b = load_frag16(wbase);            // fragment 0
#pragma unroll
        for (int nt = 0; nt < 8; ++nt) {
            v8f c0 = {}, c1 = {};
#pragma unroll
            for (int kt = 0; kt < 4; ++kt) {
                v16h cur = b;
                const int nf = nt * 4 + kt + 1;
                if (nf < 32) b = load_frag16(wbase + (size_t)nf * 32 * 16);
                c0 = __builtin_amdgcn_wmma_f32_16x16x32_f16(false, A2[0][kt], false, cur,
                                                            (short)0, c0, false, false);
                c1 = __builtin_amdgcn_wmma_f32_16x16x32_f16(false, A2[1][kt], false, cur,
                                                            (short)0, c1, false, false);
            }
            const int n = nt * 16 + lr;
            const float b2n = b2[n];
#pragma unroll
            for (int r = 0; r < 8; ++r) {
                float v0 = fmaxf(c0[r] + b2n, 0.f) * dinv[0][r];
                float v1 = fmaxf(c1[r] + b2n, 0.f) * dinv[1][r];
                atomicAdd(&aggr[(size_t)drow[0][r] * DF + n], v0);
                atomicAdd(&aggr[(size_t)drow[1][r] * DF + n], v1);
            }
        }
    }
}

// ---------- node kernel: one wave32 per 32-node tile ----------
__global__ __launch_bounds__(32 * WPB, 1)
void node_kernel(_Float16* __restrict__ xh,           // in: x f16, out: new x f16
                 const _Float16* __restrict__ agh,    // aggr f16
                 const _Float16* __restrict__ w1p,    // packed nw1, NK=8
                 const float* __restrict__ b1,
                 const _Float16* __restrict__ w2p,    // packed nw2, NK=4
                 const float* __restrict__ b2,
                 float* __restrict__ xbuf,            // new x f32 (for LN)
                 int N)
{
    __shared__ __align__(16) _Float16 mt[WPB][32][DF];
    const int w    = threadIdx.x >> 5;
    const int lane = threadIdx.x & 31;
    const int tile = blockIdx.x * WPB + w;
    const int t0   = tile * 32;
    if (t0 >= N) return;
    const int hi = lane >> 4;
    const int lr = lane & 15;

    v16h A[2][8];
#pragma unroll
    for (int s = 0; s < 2; ++s) {
        const int node = min(t0 + s * 16 + lr, N - 1);
        const _Float16* pi = xh + (size_t)node * DF;
        const _Float16* pj = agh + (size_t)node * DF;
#pragma unroll
        for (int kt = 0; kt < 8; ++kt) {
            const int k0 = kt * 32;
            const _Float16* base = (k0 < DF) ? (pi + k0) : (pj + (k0 - DF));
            v8h lo = *(const v8h*)(base + hi * 8);
            v8h hh = *(const v8h*)(base + 16 + hi * 8);
#pragma unroll
            for (int i = 0; i < 8; ++i) { A[s][kt][i] = lo[i]; A[s][kt][8 + i] = hh[i]; }
        }
    }

    // GEMM1: h = relu([x|aggr] @ nw1 + nb1), pipelined B stream
    {
        const _Float16* wbase = w1p + (size_t)lane * 16;
        v16h b = load_frag16(wbase);
#pragma unroll
        for (int nt = 0; nt < 8; ++nt) {
            v8f c0 = {}, c1 = {};
#pragma unroll
            for (int kt = 0; kt < 8; ++kt) {
                v16h cur = b;
                const int nf = nt * 8 + kt + 1;
                if (nf < 64) b = load_frag16(wbase + (size_t)nf * 32 * 16);
                c0 = __builtin_amdgcn_wmma_f32_16x16x32_f16(false, A[0][kt], false, cur,
                                                            (short)0, c0, false, false);
                c1 = __builtin_amdgcn_wmma_f32_16x16x32_f16(false, A[1][kt], false, cur,
                                                            (short)0, c1, false, false);
            }
            const int n = nt * 16 + lr;
            const float b1n = b1[n];
#pragma unroll
            for (int r = 0; r < 8; ++r) {
                mt[w][r + 8 * hi][n]      = (_Float16)fmaxf(c0[r] + b1n, 0.f);
                mt[w][16 + r + 8 * hi][n] = (_Float16)fmaxf(c1[r] + b1n, 0.f);
            }
        }
    }
    asm volatile("" ::: "memory");

    v16h A2[2][4];
#pragma unroll
    for (int s = 0; s < 2; ++s)
#pragma unroll
        for (int kt = 0; kt < 4; ++kt) {
            const _Float16* mrow = &mt[w][s * 16 + lr][kt * 32 + hi * 8];
            v8h lo = *(const v8h*)(mrow);
            v8h hh = *(const v8h*)(mrow + 16);
#pragma unroll
            for (int i = 0; i < 8; ++i) { A2[s][kt][i] = lo[i]; A2[s][kt][8 + i] = hh[i]; }
        }

    // GEMM2: x = relu(h @ nw2 + nb2), pipelined B stream
    {
        const _Float16* wbase = w2p + (size_t)lane * 16;
        v16h b = load_frag16(wbase);
#pragma unroll
        for (int nt = 0; nt < 8; ++nt) {
            v8f c0 = {}, c1 = {};
#pragma unroll
            for (int kt = 0; kt < 4; ++kt) {
                v16h cur = b;
                const int nf = nt * 4 + kt + 1;
                if (nf < 32) b = load_frag16(wbase + (size_t)nf * 32 * 16);
                c0 = __builtin_amdgcn_wmma_f32_16x16x32_f16(false, A2[0][kt], false, cur,
                                                            (short)0, c0, false, false);
                c1 = __builtin_amdgcn_wmma_f32_16x16x32_f16(false, A2[1][kt], false, cur,
                                                            (short)0, c1, false, false);
            }
            const int n = nt * 16 + lr;
            const float b2n = b2[n];
#pragma unroll
            for (int r = 0; r < 8; ++r) {
#pragma unroll
                for (int s = 0; s < 2; ++s) {
                    float v = fmaxf((s ? c1[r] : c0[r]) + b2n, 0.f);
                    int nd = t0 + s * 16 + r + 8 * hi;
                    if (nd < N) {
                        xbuf[(size_t)nd * DF + n] = v;
                        xh[(size_t)nd * DF + n] = (_Float16)v;
                    }
                }
            }
        }
    }
}

// ---------- LayerNorm: one wave32 per node ----------
__global__ void ln_kernel(const float* __restrict__ x, const float* __restrict__ gamma,
                          const float* __restrict__ beta, float* __restrict__ out, int N)
{
    int wv   = (blockIdx.x * blockDim.x + threadIdx.x) >> 5;
    int lane = threadIdx.x & 31;
    if (wv >= N) return;
    const float* row = x + (size_t)wv * DF;
    float v[4], s = 0.f, sq = 0.f;
#pragma unroll
    for (int i = 0; i < 4; ++i) {
        v[i] = row[lane * 4 + i];
        s += v[i]; sq += v[i] * v[i];
    }
#pragma unroll
    for (int m = 16; m >= 1; m >>= 1) {
        s  += __shfl_xor(s,  m, 32);
        sq += __shfl_xor(sq, m, 32);
    }
    float mean = s * (1.f / DF);
    float var  = sq * (1.f / DF) - mean * mean;
    float inv  = rsqrtf(var + 1e-5f);
    float* orow = out + (size_t)wv * DF;
#pragma unroll
    for (int i = 0; i < 4; ++i) {
        int k = lane * 4 + i;
        orow[k] = (v[i] - mean) * inv * gamma[k] + beta[k];
    }
}

// ---------- host ----------
extern "C" void kernel_launch(void* const* d_in, const int* in_sizes, int n_in,
                              void* d_out, int out_size, void* d_ws, size_t ws_size,
                              hipStream_t stream) {
    (void)n_in; (void)out_size; (void)ws_size;
    const float*     x     = (const float*)d_in[0];
    const long long* ei    = (const long long*)d_in[1];
    const float*     ewgt  = (const float*)d_in[2];
    const float*     ew1   = (const float*)d_in[3];   // [3][257][128]
    const float*     eb1   = (const float*)d_in[4];
    const float*     ew2   = (const float*)d_in[5];   // [3][128][128]
    const float*     eb2   = (const float*)d_in[6];
    const float*     nw1   = (const float*)d_in[7];   // [3][256][128]
    const float*     nb1   = (const float*)d_in[8];
    const float*     nw2   = (const float*)d_in[9];
    const float*     nb2   = (const float*)d_in[10];
    const float*     gamma = (const float*)d_in[11];
    const float*     beta  = (const float*)d_in[12];
    float* out = (float*)d_out;

    const int N = in_sizes[0] / DF;
    const int E = in_sizes[2];
    const long long* srcI = ei;
    const long long* dstI = ei + E;

    // workspace carve (256B aligned)
    char* p = (char*)d_ws;
    auto carve = [&](size_t bytes) { void* r = p; p += (bytes + 255) & ~(size_t)255; return r; };
    _Float16* xh   = (_Float16*)carve((size_t)N * DF * 2);
    _Float16* agh  = (_Float16*)carve((size_t)N * DF * 2);
    float*    xbuf = (float*)carve((size_t)N * DF * 4);
    float*    aggr = (float*)carve((size_t)N * DF * 4);
    float*    cnt  = (float*)carve((size_t)N * 4);
    float*    invd = (float*)carve((size_t)N * 4);
    _Float16* w1p  = (_Float16*)carve((size_t)3 * 256 * DF * 2);
    _Float16* w2p  = (_Float16*)carve((size_t)3 * 128 * DF * 2);
    _Float16* n1p  = (_Float16*)carve((size_t)3 * 256 * DF * 2);
    _Float16* n2p  = (_Float16*)carve((size_t)3 * 128 * DF * 2);

    const int T = 256;
    init_x<<<(N * DF + T - 1) / T, T, 0, stream>>>(x, xh, xbuf, N * DF);
    zero_f32<<<(N + T - 1) / T, T, 0, stream>>>(cnt, N);
    count_kernel<<<(E + T - 1) / T, T, 0, stream>>>(dstI, cnt, E);
    invden_kernel<<<(N + T - 1) / T, T, 0, stream>>>(cnt, invd, N);

    // repack all weights to WMMA B-fragment f16 layout
    for (int l = 0; l < 3; ++l) {
        repack_b<<<(8 * 4096 + T - 1) / T, T, 0, stream>>>(ew1 + (size_t)l * 257 * DF, w1p + (size_t)l * 256 * DF, 8);
        repack_b<<<(4 * 4096 + T - 1) / T, T, 0, stream>>>(ew2 + (size_t)l * 128 * DF, w2p + (size_t)l * 128 * DF, 4);
        repack_b<<<(8 * 4096 + T - 1) / T, T, 0, stream>>>(nw1 + (size_t)l * 256 * DF, n1p + (size_t)l * 256 * DF, 8);
        repack_b<<<(4 * 4096 + T - 1) / T, T, 0, stream>>>(nw2 + (size_t)l * 128 * DF, n2p + (size_t)l * 128 * DF, 4);
    }

    const int eTiles = (E + 31) / 32;            // 32 edges per wave
    const int nTiles = (N + 31) / 32;            // 32 nodes per wave
    const int B = 32 * WPB;
    for (int l = 0; l < 3; ++l) {
        zero_f32<<<(N * DF + T - 1) / T, T, 0, stream>>>(aggr, N * DF);
        edge_kernel<<<(eTiles + WPB - 1) / WPB, B, 0, stream>>>(
            xh, srcI, dstI, ewgt,
            w1p + (size_t)l * 256 * DF,
            ew1 + (size_t)l * 257 * DF + 256 * DF,   // row 256 (edge-weight feature)
            eb1 + (size_t)l * DF,
            w2p + (size_t)l * 128 * DF,
            eb2 + (size_t)l * DF,
            invd, aggr, E);
        f32_to_f16<<<(N * DF + T - 1) / T, T, 0, stream>>>(aggr, agh, N * DF);
        node_kernel<<<(nTiles + WPB - 1) / WPB, B, 0, stream>>>(
            xh, agh,
            n1p + (size_t)l * 256 * DF, nb1 + (size_t)l * DF,
            n2p + (size_t)l * 128 * DF, nb2 + (size_t)l * DF,
            xbuf, N);
    }

    ln_kernel<<<(N + 7) / 8, T, 0, stream>>>(xbuf, gamma, beta, out, N);
}
